// LinearAttention_52407190946367
// MI455X (gfx1250) — matrix operand order repaired and checked by reading, not compiled, not run on hardware
//
#include <hip/hip_runtime.h>
#include <hip/hip_bf16.h>

// ---------------------------------------------------------------------------
// Linear attention for MI455X (gfx1250): bf16 WMMA pipeline, fp32 accumulate,
// async-to-LDS double-buffered GEMM main loops.
// ---------------------------------------------------------------------------

typedef __bf16 bf16;
typedef __attribute__((ext_vector_type(16))) __bf16 v16bf;
typedef __attribute__((ext_vector_type(8)))  __bf16 v8bf;
typedef __attribute__((ext_vector_type(8)))  float  v8f;

#define D_MODEL 1024
#define N_HEADS 16
#define HEAD_DIM 64
#define BATCH 4
#define SEQ 4096
#define MTOT (BATCH * SEQ)        // 16384 rows of x
#define EPS 1e-6f
#define LDS_PITCH 40              // 32 bf16 + 8 pad -> conflict-free ds_load_b128

// -------------------------- WMMA helpers -----------------------------------

__device__ __forceinline__ v8f wmma_bf16(v16bf a, v16bf b, v8f c) {
  return __builtin_amdgcn_wmma_f32_16x16x32_bf16(
      /*neg_a=*/false, a, /*neg_b=*/false, b,
      /*c_mod=*/(short)0, c, /*reuse_a=*/false, /*reuse_b=*/false);
}

// A fragment (16x32 MxK) from row-major memory: per ISA 7.12.2 lane half h
// holds K chunks [8h,8h+8) and [16+8h,16+8h+8).
__device__ __forceinline__ v16bf load_a_frag(const bf16* __restrict__ base, int ld) {
  const int lane = threadIdx.x & 31;
  const int r = lane & 15, hf = lane >> 4;
  const bf16* p = base + (size_t)r * ld + hf * 8;
  union { v16bf v; v8bf h[2]; } u;
  u.h[0] = *(const v8bf*)(p);
  u.h[1] = *(const v8bf*)(p + 16);
  return u.v;
}

// B fragment (32x16 KxN) from a TRANSPOSED source (N rows x K cols): lane col
// = lane&15, half h reads K chunks [8h,8h+8) and [16+8h,16+8h+8).
__device__ __forceinline__ v16bf load_bt_frag(const bf16* __restrict__ base, int ld) {
  const int lane = threadIdx.x & 31;
  const int cl = lane & 15, hf = lane >> 4;
  const bf16* p = base + (size_t)cl * ld + hf * 8;
  union { v16bf v; v8bf h[2]; } u;
  u.h[0] = *(const v8bf*)(p);
  u.h[1] = *(const v8bf*)(p + 16);
  return u.v;
}

__device__ __forceinline__ float fmap(float x) {  // elu(x) + 1
  return x > 0.f ? x + 1.f : __expf(x);
}

// CDNA5 async copy: global -> LDS, 16B per lane, tracked by ASYNCcnt.
__device__ __forceinline__ void async_copy_b128(unsigned lds_off, const bf16* gptr) {
  asm volatile("global_load_async_to_lds_b128 %0, %1, off"
               :: "v"(lds_off), "v"(gptr) : "memory");
}
__device__ __forceinline__ void wait_async0() {
  asm volatile("s_wait_asynccnt 0x0" ::: "memory");
}

// ------------------------ conversion kernels --------------------------------

__global__ void cvt_f32_bf16(const float* __restrict__ src, bf16* __restrict__ dst, int n) {
  for (int i = blockIdx.x * blockDim.x + threadIdx.x; i < n; i += gridDim.x * blockDim.x)
    dst[i] = (bf16)src[i];
}

// W (1024x1024, [k][n] fp32) -> Wt (1024x1024, [n][k] bf16), LDS-tiled.
__global__ __launch_bounds__(256) void wtrans(const float* __restrict__ src,
                                              bf16* __restrict__ dst) {
  __shared__ bf16 t[32][33];
  const int bn = blockIdx.x * 32, bk = blockIdx.y * 32;
  const int tx = threadIdx.x, ty = threadIdx.y;   // block (32, 8)
#pragma unroll
  for (int i = ty; i < 32; i += 8)
    t[i][tx] = (bf16)src[(size_t)(bk + i) * D_MODEL + bn + tx];   // (k, n)
  __syncthreads();
#pragma unroll
  for (int i = ty; i < 32; i += 8)
    dst[(size_t)(bn + i) * D_MODEL + bk + tx] = t[tx][i];          // (n, k)
}

// --------------- unified LDS-staged GEMM: C = Arows x Brows^T ---------------
// Arows: (Mdim x 1024) row-major bf16 (tile rows = WMMA M)
// Brows: (Ndim x 1024) row-major bf16 (tile rows = WMMA N)
// EPI 0: Q  proj  -> fmap, bf16 [b*h][s][64]          (A=X,  B=WqT)
// EPI 1: out proj -> fp32 [m][1024] + bias[n]         (A=attn, B=WoT)
// EPI 2: K^T proj -> fmap, bf16 [b*h][d][S]           (A=WkT, B=X)
// EPI 3: V^T proj ->        bf16 [b*h][d][S]          (A=WvT, B=X)
template <int EPI>
__global__ __launch_bounds__(256) void gemm_lds(const bf16* __restrict__ Arows,
                                                const bf16* __restrict__ Brows,
                                                const float* __restrict__ bias,
                                                void* __restrict__ dst) {
  __shared__ bf16 As[2][128 * LDS_PITCH];   // 128-row x 32-K tile, padded
  __shared__ bf16 Bs[2][256 * LDS_PITCH];   // 256-row x 32-K tile, padded

  const int tid = threadIdx.x;
  const int wid = tid >> 5, lane = tid & 31;
  const int mBase = blockIdx.x * 128, nBase = blockIdx.y * 256;
  const int wm = (wid & 1) * 64, wn = (wid >> 1) * 64;

  auto stage = [&](int buf, int kk) {
#pragma unroll
    for (int c = 0; c < 2; ++c) {           // A: 512 x 16B chunks
      const int cidx = tid + c * 256;
      const int row = cidx >> 2, cc = cidx & 3;
      async_copy_b128((unsigned)(size_t)&As[buf][row * LDS_PITCH + cc * 8],
                      Arows + (size_t)(mBase + row) * D_MODEL + kk + cc * 8);
    }
#pragma unroll
    for (int c = 0; c < 4; ++c) {           // B: 1024 x 16B chunks
      const int cidx = tid + c * 256;
      const int row = cidx >> 2, cc = cidx & 3;
      async_copy_b128((unsigned)(size_t)&Bs[buf][row * LDS_PITCH + cc * 8],
                      Brows + (size_t)(nBase + row) * D_MODEL + kk + cc * 8);
    }
  };

  stage(0, 0);
  wait_async0();
  __syncthreads();

  v8f c[4][4] = {};
  const int r = lane & 15, hf = lane >> 4;
  int buf = 0;
  for (int kk = 0; kk < D_MODEL; kk += 32) {
    if (kk + 32 < D_MODEL) stage(buf ^ 1, kk + 32);   // overlap next stage

    v16bf a[4], b[4];
#pragma unroll
    for (int i = 0; i < 4; ++i) {
      const bf16* p = &As[buf][(wm + 16 * i + r) * LDS_PITCH + hf * 8];
      union { v16bf v; v8bf h[2]; } u;
      u.h[0] = *(const v8bf*)p;
      u.h[1] = *(const v8bf*)(p + 16);
      a[i] = u.v;
    }
#pragma unroll
    for (int j = 0; j < 4; ++j) {
      const bf16* p = &Bs[buf][(wn + 16 * j + r) * LDS_PITCH + hf * 8];
      union { v16bf v; v8bf h[2]; } u;
      u.h[0] = *(const v8bf*)p;
      u.h[1] = *(const v8bf*)(p + 16);
      b[j] = u.v;
    }
#pragma unroll
    for (int i = 0; i < 4; ++i)
#pragma unroll
      for (int j = 0; j < 4; ++j)
        c[i][j] = wmma_bf16(a[i], b[j], c[i][j]);

    wait_async0();        // own stage complete before barrier
    __syncthreads();      // all waves done reading buf / staging buf^1
    buf ^= 1;
  }

  const int cn = lane & 15, ch = lane >> 4;
#pragma unroll
  for (int i = 0; i < 4; ++i)
#pragma unroll
    for (int j = 0; j < 4; ++j)
#pragma unroll
      for (int e = 0; e < 8; ++e) {
        const int mi = mBase + wm + 16 * i + e + 8 * ch;   // A-row index
        const int nj = nBase + wn + 16 * j + cn;           // B-row index
        float v = c[i][j][e];
        if (EPI == 0) {
          v = fmap(v + bias[nj]);
          const int b = mi >> 12, s = mi & 4095, h = nj >> 6, d = nj & 63;
          ((bf16*)dst)[(((size_t)(b * N_HEADS + h) * SEQ) + s) * HEAD_DIM + d] = (bf16)v;
        } else if (EPI == 1) {
          ((float*)dst)[(size_t)mi * D_MODEL + nj] = v + bias[nj];
        } else {
          v += bias[mi];
          if (EPI == 2) v = fmap(v);
          const int b = nj >> 12, s = nj & 4095, h = mi >> 6, d = mi & 63;
          ((bf16*)dst)[(((size_t)(b * N_HEADS + h) * HEAD_DIM) + d) * SEQ + s] = (bf16)v;
        }
      }
}

// ---------------- kv = K^T V (+ k_sum) per (b,h), over S --------------------
// kt/vt: [bh][64][S]; output kvT_ext: [bh][80][64] bf16, rows 64..79 = k_sum.
// No conditional WMMA: all waves compute a k_sum tile, waves 0..3 store it.
__global__ __launch_bounds__(256) void kv_kernel(const bf16* __restrict__ kt,
                                                 const bf16* __restrict__ vt,
                                                 bf16* __restrict__ kvt) {
  const int bh = blockIdx.x;
  const bf16* K = kt + (size_t)bh * HEAD_DIM * SEQ;
  const bf16* V = vt + (size_t)bh * HEAD_DIM * SEQ;
  bf16* KV = kvt + (size_t)bh * 80 * HEAD_DIM;

  const int wid = threadIdx.x >> 5;
  const int tm = wid >> 1;            // kv row tile (d)
  const int tn0 = (wid & 1) * 2;      // kv col tiles (e, e+1)
  const int tms = wid & 3;            // k_sum row tile (all waves compute)

  union { v16bf v; __bf16 e[16]; } ones;
#pragma unroll
  for (int i = 0; i < 16; ++i) ones.e[i] = (__bf16)1.0f;

  v8f c0 = {}, c1 = {}, cs = {};
  // software pipeline: loads for step n+32 issued before WMMAs of step n
  v16bf a  = load_a_frag(K + (size_t)(tm * 16) * SEQ, SEQ);
  v16bf b0 = load_bt_frag(V + (size_t)(tn0 * 16) * SEQ, SEQ);
  v16bf b1 = load_bt_frag(V + (size_t)((tn0 + 1) * 16) * SEQ, SEQ);
  v16bf as = load_a_frag(K + (size_t)(tms * 16) * SEQ, SEQ);
  for (int n = 0; n < SEQ; n += 32) {
    const int nn = (n + 32 < SEQ) ? (n + 32) : n;
    v16bf na  = load_a_frag(K + (size_t)(tm * 16) * SEQ + nn, SEQ);
    v16bf nb0 = load_bt_frag(V + (size_t)(tn0 * 16) * SEQ + nn, SEQ);
    v16bf nb1 = load_bt_frag(V + (size_t)((tn0 + 1) * 16) * SEQ + nn, SEQ);
    v16bf nas = load_a_frag(K + (size_t)(tms * 16) * SEQ + nn, SEQ);
    c0 = wmma_bf16(a, b0, c0);
    c1 = wmma_bf16(a, b1, c1);
    cs = wmma_bf16(as, ones.v, cs);
    a = na; b0 = nb0; b1 = nb1; as = nas;
  }

  const int lane = threadIdx.x & 31;
  const int cn = lane & 15, ch = lane >> 4;
#pragma unroll
  for (int e = 0; e < 8; ++e) {
    const int d = tm * 16 + e + 8 * ch;
    KV[(size_t)(tn0 * 16 + cn) * HEAD_DIM + d]       = (bf16)c0[e];
    KV[(size_t)((tn0 + 1) * 16 + cn) * HEAD_DIM + d] = (bf16)c1[e];
  }
  if (wid < 4) {   // store-only branch; no WMMA under modified EXEC
#pragma unroll
    for (int e = 0; e < 8; ++e) {
      const int d = tms * 16 + e + 8 * ch;   // rows 64..79 all hold k_sum
      KV[(size_t)(64 + cn) * HEAD_DIM + d] = (bf16)cs[e];
    }
  }
}

// ------------- out = (q @ kv) / (q . k_sum + eps), per (b,h) ----------------
__global__ __launch_bounds__(256) void attn_kernel(const bf16* __restrict__ q,
                                                   const bf16* __restrict__ kvt,
                                                   bf16* __restrict__ attn) {
  const int bh = blockIdx.x;
  const int b = bh >> 4, h = bh & 15;
  const bf16* Q  = q + (size_t)bh * SEQ * HEAD_DIM;
  const bf16* KV = kvt + (size_t)bh * 80 * HEAD_DIM;

  const int wid = threadIdx.x >> 5;
  const int m0 = blockIdx.y * 512 + wid * 64;

  v8f c[4][5] = {};
#pragma unroll
  for (int kk = 0; kk < HEAD_DIM; kk += 32) {
    v16bf a[4], bb[5];
#pragma unroll
    for (int i = 0; i < 4; ++i)
      a[i] = load_a_frag(Q + (size_t)(m0 + 16 * i) * HEAD_DIM + kk, HEAD_DIM);
#pragma unroll
    for (int j = 0; j < 5; ++j)   // j==4: normalizer tile (k_sum rows)
      bb[j] = load_bt_frag(KV + (size_t)(j * 16) * HEAD_DIM + kk, HEAD_DIM);
#pragma unroll
    for (int i = 0; i < 4; ++i)
#pragma unroll
      for (int j = 0; j < 5; ++j)
        c[i][j] = wmma_bf16(a[i], bb[j], c[i][j]);
  }

  const int lane = threadIdx.x & 31;
  const int cn = lane & 15, ch = lane >> 4;
#pragma unroll
  for (int i = 0; i < 4; ++i)
#pragma unroll
    for (int e = 0; e < 8; ++e) {
      const float norm = c[i][4][e] + EPS;   // same value in every column
      const int s = m0 + 16 * i + e + 8 * ch;
#pragma unroll
      for (int j = 0; j < 4; ++j) {
        attn[((size_t)(b * SEQ + s)) * D_MODEL + h * HEAD_DIM + j * 16 + cn] =
            (bf16)(c[i][j][e] / norm);
      }
    }
}

// ----------------------------- launcher -------------------------------------

extern "C" void kernel_launch(void* const* d_in, const int* in_sizes, int n_in,
                              void* d_out, int out_size, void* d_ws, size_t ws_size,
                              hipStream_t stream) {
  const float* x  = (const float*)d_in[0];
  const float* Wq = (const float*)d_in[1];
  const float* bq = (const float*)d_in[2];
  const float* Wk = (const float*)d_in[3];
  const float* bk = (const float*)d_in[4];
  const float* Wv = (const float*)d_in[5];
  const float* bv = (const float*)d_in[6];
  const float* Wo = (const float*)d_in[7];
  const float* bo = (const float*)d_in[8];
  float* out = (float*)d_out;

  char* ws = (char*)d_ws;
  size_t off = 0;
  bf16* Xbf = (bf16*)(ws + off); off += (size_t)MTOT * D_MODEL * 2;   // aliased as attn later
  bf16* WqT = (bf16*)(ws + off); off += (size_t)D_MODEL * D_MODEL * 2;
  bf16* WkT = (bf16*)(ws + off); off += (size_t)D_MODEL * D_MODEL * 2;
  bf16* WvT = (bf16*)(ws + off); off += (size_t)D_MODEL * D_MODEL * 2;
  bf16* WoT = (bf16*)(ws + off); off += (size_t)D_MODEL * D_MODEL * 2;
  bf16* qf  = (bf16*)(ws + off); off += (size_t)MTOT * D_MODEL * 2;   // [bh][s][64]
  bf16* ktb = (bf16*)(ws + off); off += (size_t)MTOT * D_MODEL * 2;   // [bh][64][S]
  bf16* vtb = (bf16*)(ws + off); off += (size_t)MTOT * D_MODEL * 2;   // [bh][64][S]
  bf16* kvt = (bf16*)(ws + off); off += (size_t)BATCH * N_HEADS * 80 * HEAD_DIM * 2;

  // 1) conversions
  cvt_f32_bf16<<<2048, 256, 0, stream>>>(x, Xbf, MTOT * D_MODEL);
  dim3 tg(32, 32), tb(32, 8);
  wtrans<<<tg, tb, 0, stream>>>(Wq, WqT);
  wtrans<<<tg, tb, 0, stream>>>(Wk, WkT);
  wtrans<<<tg, tb, 0, stream>>>(Wv, WvT);
  wtrans<<<tg, tb, 0, stream>>>(Wo, WoT);

  // 2) Q projection (+feature map)
  gemm_lds<0><<<dim3(MTOT / 128, D_MODEL / 256), 256, 0, stream>>>(Xbf, WqT, bq, (void*)qf);

  // 3) K, V projections written transposed [bh][64][S]
  gemm_lds<2><<<dim3(D_MODEL / 128, MTOT / 256), 256, 0, stream>>>(WkT, Xbf, bk, (void*)ktb);
  gemm_lds<3><<<dim3(D_MODEL / 128, MTOT / 256), 256, 0, stream>>>(WvT, Xbf, bv, (void*)vtb);

  // 4) kv + k_sum per (b,h)
  kv_kernel<<<BATCH * N_HEADS, 256, 0, stream>>>(ktb, vtb, kvt);

  // 5) normalized attention output -> bf16 [B][S][D] (reuses Xbf storage)
  attn_kernel<<<dim3(BATCH * N_HEADS, SEQ / 512), 256, 0, stream>>>(qf, kvt, Xbf);

  // 6) output projection, fp32
  gemm_lds<1><<<dim3(MTOT / 128, D_MODEL / 256), 256, 0, stream>>>(Xbf, WoT, bo, (void*)out);
}
